// SCAB_45028437131674
// MI455X (gfx1250) — matrix operand order repaired and checked by reading, not compiled, not run on hardware
//
#include <hip/hip_runtime.h>
#include <math.h>

// Problem constants (from reference): B=8, C=384, H=W=64, heads=8, Cph=48
#define BDIM  8
#define CDIM  384
#define HDIM  64
#define WDIM  64
#define SDIM  4096
#define HEADS 8
#define CPH   48

typedef __attribute__((ext_vector_type(16))) _Float16 v16h;
typedef __attribute__((ext_vector_type(8)))  _Float16 v8h;
typedef __attribute__((ext_vector_type(8)))  float    v8f;

union F16x16 { v16h v; v8h h[2]; };

__device__ __forceinline__ float ldf(const float* p)    { return *p; }
__device__ __forceinline__ float ldf(const _Float16* p) { return (float)*p; }
__device__ __forceinline__ void  stf(float* p, float v)    { *p = v; }
__device__ __forceinline__ void  stf(_Float16* p, float v) { *p = (_Float16)v; }

// ---------------------------------------------------------------------------
// Pointwise-conv GEMM, full-K staging:
//   Y[b,m,n] = sum_k W[m,k] * X[b,k,n]      (K = CDIM = 384 fixed)
// Each block owns a 64-column N-slice and stages X[b, 0:384, n0:n0+64] in LDS
// ONCE (f16, [n][k], ~51 KB) -> every activation element is read from HBM
// exactly once per GEMM. It then loops over M in 128-row blocks, staging the
// (L2-resident) weight tile per K-step.  8 waves: wave w = M-tile w, all 4
// N-tiles (4 x v8f accumulators).  M-guard supports M = 384 / 768 / 192.
// ---------------------------------------------------------------------------
#define LXS 408   // lX row stride (f16): 384 + 24 pad, keeps 16B alignment
#define LAS 40    // lA row stride (f16): 32 + 8 pad

template <typename TIN, typename TOUT, bool BIAS_RELU>
__global__ __launch_bounds__(256) void scab_gemm1x1_fullk(
    const float* __restrict__ Wmat, const TIN* __restrict__ X,
    const float* __restrict__ bias, TOUT* __restrict__ Y, int Mdim)
{
  __shared__ _Float16 lX[64 * LXS];    // [n][k] full-K activation strip
  __shared__ _Float16 lA[128 * LAS];   // [m][k] weight tile per K-step

  const int tid = threadIdx.x;
  const int b   = blockIdx.y;
  const int n0  = blockIdx.x * 64;

  // ---- stage X strip once: 64(n) x 384(k), 4 consecutive n per thread ----
  for (int j = 0; j < 24; ++j) {
    int i = tid + 256 * j;                 // 0 .. 6143
    int k = i >> 4, n4 = (i & 15) * 4;
    const TIN* src = X + ((size_t)b * CDIM + k) * SDIM + n0 + n4;
    lX[(n4 + 0) * LXS + k] = (_Float16)ldf(src + 0);
    lX[(n4 + 1) * LXS + k] = (_Float16)ldf(src + 1);
    lX[(n4 + 2) * LXS + k] = (_Float16)ldf(src + 2);
    lX[(n4 + 3) * LXS + k] = (_Float16)ldf(src + 3);
  }
  __syncthreads();

  const int lane = tid & 31, w = tid >> 5;
  const bool hi = lane >= 16;
  const int l16 = lane & 15;
  const int aoff = hi ? 8 : 0, boff = hi ? 8 : 0;

  for (int m0 = 0; m0 < Mdim; m0 += 128) {
    v8f acc[4];
    for (int t = 0; t < 4; ++t)
      for (int r = 0; r < 8; ++r) acc[t][r] = 0.f;

    for (int k0 = 0; k0 < CDIM; k0 += 32) {
      // stage A tile 128x32 (f32 -> f16), zero rows past Mdim
      for (int j = 0; j < 16; ++j) {
        int i = tid + 256 * j; int m = i >> 5, k = i & 31;
        int mg = m0 + m;
        lA[m * LAS + k] = (mg < Mdim)
            ? (_Float16)Wmat[(size_t)mg * CDIM + k0 + k] : (_Float16)0.f;
      }
      if (k0 + 32 < CDIM)
        __builtin_prefetch(Wmat + (size_t)(m0 + (tid >> 1)) * CDIM + k0 + 32, 0, 0);
      __syncthreads();

      // A fragment: lane<16 holds K{0..7,16..23}; lane>=16 K{8..15,24..31}
      F16x16 a;
      const _Float16* ar = &lA[(w * 16 + l16) * LAS + aoff];
      a.h[0] = *(const v8h*)(ar);
      a.h[1] = *(const v8h*)(ar + 16);
      for (int t = 0; t < 4; ++t) {
        // B fragment: lane<16 holds K{0..7,16..23} of its column; lane>=16 K{8..15,24..31}
        F16x16 bb;
        const _Float16* br = &lX[(t * 16 + l16) * LXS + k0 + boff];
        bb.h[0] = *(const v8h*)(br);
        bb.h[1] = *(const v8h*)(br + 16);
        acc[t] = __builtin_amdgcn_wmma_f32_16x16x32_f16(false, a.v, false, bb.v,
                                                        (short)0, acc[t], false, false);
      }
      __syncthreads();   // lA rewritten next K-step
    }

    // C layout: VGPR r -> M = r (lanes 0..15) / 8+r (lanes 16..31); N = lane%16
    int mbase = m0 + w * 16 + (hi ? 8 : 0);
    for (int t = 0; t < 4; ++t) {
      int ncol = n0 + t * 16 + l16;
      for (int r = 0; r < 8; ++r) {
        int m = mbase + r;
        if (m < Mdim) {
          float v = acc[t][r];
          if constexpr (BIAS_RELU) v = fmaxf(v + bias[m], 0.f);
          stf(Y + ((size_t)b * Mdim + m) * SDIM + ncol, v);
        }
      }
    }
  }
}

// ---------------------------------------------------------------------------
// Depthwise 3x3 SAME, groups=C (f16 in -> f16 out, f32 accumulate)
// ---------------------------------------------------------------------------
__global__ void scab_dwconv3x3(const _Float16* __restrict__ in,
                               const float* __restrict__ w9,
                               _Float16* __restrict__ out, int Cc, int total)
{
  int idx = blockIdx.x * 256 + threadIdx.x;
  if (idx >= total) return;
  int p = idx & (SDIM - 1);
  int c = (idx >> 12) % Cc;
  int y = p >> 6, x = p & 63;
  const _Float16* base = in + (size_t)(idx - p);   // channel-plane start
  float acc = 0.f;
  for (int dy = 0; dy < 3; ++dy) {
    int yy = y + dy - 1;
    if (yy < 0 || yy >= HDIM) continue;
    for (int dx = 0; dx < 3; ++dx) {
      int xx = x + dx - 1;
      if (xx < 0 || xx >= WDIM) continue;
      acc += (float)base[(yy << 6) + xx] * w9[c * 9 + dy * 3 + dx];
    }
  }
  out[idx] = (_Float16)acc;
}

// ---------------------------------------------------------------------------
// Inverse L2 norm per (b, channel) row over S=4096.
// ---------------------------------------------------------------------------
__global__ void scab_rownorm(const _Float16* __restrict__ rows, int batchStride,
                             float* __restrict__ inv)
{
  __shared__ float red[256];
  int r = blockIdx.x;                       // 0 .. B*384-1
  int b = r / CDIM, c = r % CDIM;
  const _Float16* p = rows + ((size_t)b * batchStride + c) * SDIM;
  float s = 0.f;
  for (int j = 0; j < 16; ++j) {
    float v = (float)p[threadIdx.x + 256 * j];
    s += v * v;
  }
  red[threadIdx.x] = s; __syncthreads();
  for (int o = 128; o > 0; o >>= 1) {
    if (threadIdx.x < o) red[threadIdx.x] += red[threadIdx.x + o];
    __syncthreads();
  }
  if (threadIdx.x == 0) inv[r] = 1.f / fmaxf(sqrtf(red[0]), 1e-12f);
}

// ---------------------------------------------------------------------------
// Gate: sigmoid(Wg2 . g1 + bg2) partial sums (deterministic 2-pass reduction)
// ---------------------------------------------------------------------------
__global__ void scab_gate_partial(const _Float16* __restrict__ g1,
                                  const float* __restrict__ Wg2,
                                  const float* __restrict__ bg2,
                                  float* __restrict__ partials)
{
  __shared__ float red[256];
  int sg = blockIdx.x * 256 + threadIdx.x;  // 0 .. B*S-1 (32768)
  int b = sg >> 12, s = sg & (SDIM - 1);
  const _Float16* p = g1 + (size_t)b * 192 * SDIM + s;
  float dot = 0.f;
  for (int c = 0; c < 192; ++c) dot += (float)p[(size_t)c * SDIM] * Wg2[c];
  float v = 1.f / (1.f + expf(-(dot + bg2[0])));
  red[threadIdx.x] = v; __syncthreads();
  for (int o = 128; o > 0; o >>= 1) {
    if (threadIdx.x < o) red[threadIdx.x] += red[threadIdx.x + o];
    __syncthreads();
  }
  if (threadIdx.x == 0) partials[blockIdx.x] = red[0];
}

__global__ void scab_gate_finalize(const float* __restrict__ partials,
                                   int* __restrict__ dkp)
{
  __shared__ float red[128];
  red[threadIdx.x] = partials[threadIdx.x];
  __syncthreads();
  for (int o = 64; o > 0; o >>= 1) {
    if (threadIdx.x < o) red[threadIdx.x] += red[threadIdx.x + o];
    __syncthreads();
  }
  if (threadIdx.x == 0) {
    float mean = red[0] / (float)(BDIM * SDIM);
    int dk = (int)floorf((float)CPH * mean);
    if (dk < 1) dk = 1;
    if (dk > CPH) dk = CPH;
    *dkp = dk;
  }
}

// ---------------------------------------------------------------------------
// attn[b,h,c,d] = invq[c]*invk[d]*temp[h] * sum_s q[c,s]*k[d,s]
// 9 waves per block (288 thr): wave -> one 16x16 tile of the 48x48 output.
// K=4096 loop of WMMAs fed straight from global (rows contiguous in s).
// ---------------------------------------------------------------------------
__global__ __launch_bounds__(288) void scab_attn_qk(
    const _Float16* __restrict__ qf, const _Float16* __restrict__ kvf,
    const float* __restrict__ invq, const float* __restrict__ invk,
    const float* __restrict__ temp, float* __restrict__ attn)
{
  int bh = blockIdx.x; int b = bh >> 3, h = bh & 7;
  int w = threadIdx.x >> 5, lane = threadIdx.x & 31;
  int mt = w / 3, nt = w % 3;
  bool hi = lane >= 16; int l16 = lane & 15;
  const _Float16* qrow = qf  + ((size_t)b * CDIM       + h * CPH + mt * 16 + l16) * SDIM;
  const _Float16* krow = kvf + ((size_t)b * (2 * CDIM) + h * CPH + nt * 16 + l16) * SDIM;
  const int aoff = hi ? 8 : 0, boff = hi ? 8 : 0;
  v8f acc; for (int r = 0; r < 8; ++r) acc[r] = 0.f;
  for (int s0 = 0; s0 < SDIM; s0 += 32) {
    F16x16 a, bb;
    a.h[0]  = *(const v8h*)(qrow + s0 + aoff);
    a.h[1]  = *(const v8h*)(qrow + s0 + 16 + aoff);
    bb.h[0] = *(const v8h*)(krow + s0 + boff);
    bb.h[1] = *(const v8h*)(krow + s0 + boff + 16);
    acc = __builtin_amdgcn_wmma_f32_16x16x32_f16(false, a.v, false, bb.v,
                                                 (short)0, acc, false, false);
  }
  int n = nt * 16 + l16;
  float sk = invk[b * CDIM + h * CPH + n];
  float tp = temp[h];
  for (int r = 0; r < 8; ++r) {
    int m = mt * 16 + (hi ? 8 : 0) + r;
    attn[((size_t)bh * CPH + m) * CPH + n] =
        acc[r] * invq[b * CDIM + h * CPH + m] * sk * tp;
  }
}

// ---------------------------------------------------------------------------
// Per-row dynamic-topk mask + softmax. thr = dk-th largest (rank counting,
// tie-exact vs sort+slice). Emits 48x64 zero-padded f16 for the next GEMM.
// ---------------------------------------------------------------------------
__global__ void scab_topk_softmax(const float* __restrict__ attn,
                                  const int* __restrict__ dkp,
                                  _Float16* __restrict__ attnb)
{
  int t = threadIdx.x;
  if (t >= CPH) return;
  const float* row = attn + ((size_t)blockIdx.x * CPH + t) * CPH;
  float a[CPH];
  for (int j = 0; j < CPH; ++j) a[j] = row[j];
  int dk = *dkp;
  float thr = -3.4e38f;
  for (int i = 0; i < CPH; ++i) {
    int g = 0, e = 0;
    for (int j = 0; j < CPH; ++j) { g += (a[j] > a[i]); e += (a[j] == a[i]); }
    if (g < dk && dk <= g + e) thr = a[i];
  }
  float m = -3.4e38f;
  for (int j = 0; j < CPH; ++j) if (a[j] >= thr && a[j] > m) m = a[j];
  float ex[CPH]; float sum = 0.f;
  for (int j = 0; j < CPH; ++j) {
    float e = (a[j] >= thr) ? expf(a[j] - m) : 0.f;
    ex[j] = e; sum += e;
  }
  float rs = 1.f / sum;
  _Float16* orow = attnb + ((size_t)blockIdx.x * CPH + t) * 64;
  for (int j = 0; j < CPH; ++j) orow[j] = (_Float16)(ex[j] * rs);
  for (int j = CPH; j < 64; ++j) orow[j] = (_Float16)0.f;
}

// ---------------------------------------------------------------------------
// out[b,h,c,s] = sum_d attn[c,d] * v[d,s].  K padded 48->64 (A cols + LDS
// v rows zero-filled). v tile transposed into LDS as [n][d] for B-frag loads.
// ---------------------------------------------------------------------------
__global__ __launch_bounds__(256) void scab_attn_v(
    const _Float16* __restrict__ attnb, const _Float16* __restrict__ kvf,
    _Float16* __restrict__ outf)
{
  __shared__ _Float16 lV[128 * 80];   // [n][d] stride 80
  int tid = threadIdx.x;
  int n0 = blockIdx.x * 128;
  int bh = blockIdx.y; int b = bh >> 3, h = bh & 7;
  const _Float16* vbase = kvf + ((size_t)b * (2 * CDIM) + CDIM + h * CPH) * SDIM;
  for (int j = 0; j < 24; ++j) {                 // 48 x 128 v tile, transposed
    int i = tid + 256 * j; int d = i >> 7, n = i & 127;
    lV[n * 80 + d] = vbase[(size_t)d * SDIM + n0 + n];
  }
  for (int j = 0; j < 8; ++j) {                  // zero-pad rows d = 48..63
    int i = tid + 256 * j; int n = i >> 4, d = 48 + (i & 15);
    lV[n * 80 + d] = (_Float16)0.f;
  }
  __syncthreads();

  int w = tid >> 5, lane = tid & 31;
  bool hi = lane >= 16; int l16 = lane & 15;
  v8f acc[3];
  for (int t = 0; t < 3; ++t) for (int r = 0; r < 8; ++r) acc[t][r] = 0.f;
  const _Float16* arow_base = attnb + (size_t)bh * CPH * 64;
  for (int ks = 0; ks < 64; ks += 32) {
    F16x16 bb;
    const _Float16* br = &lV[(w * 16 + l16) * 80 + ks + (hi ? 8 : 0)];
    bb.h[0] = *(const v8h*)(br);
    bb.h[1] = *(const v8h*)(br + 16);
    for (int mt = 0; mt < 3; ++mt) {
      F16x16 a;
      const _Float16* ar = arow_base + (mt * 16 + l16) * 64 + ks + (hi ? 8 : 0);
      a.h[0] = *(const v8h*)(ar);
      a.h[1] = *(const v8h*)(ar + 16);
      acc[mt] = __builtin_amdgcn_wmma_f32_16x16x32_f16(false, a.v, false, bb.v,
                                                       (short)0, acc[mt], false, false);
    }
  }
  for (int mt = 0; mt < 3; ++mt) {
    for (int r = 0; r < 8; ++r) {
      int m = mt * 16 + (hi ? 8 : 0) + r;
      int c = h * CPH + m;
      outf[((size_t)b * CDIM + c) * SDIM + n0 + w * 16 + l16] = (_Float16)acc[mt][r];
    }
  }
}

// ---------------------------------------------------------------------------
extern "C" void kernel_launch(void* const* d_in, const int* in_sizes, int n_in,
                              void* d_out, int out_size, void* d_ws, size_t ws_size,
                              hipStream_t stream)
{
  (void)in_sizes; (void)n_in; (void)out_size; (void)ws_size;
  const float* x     = (const float*)d_in[0];
  const float* y     = (const float*)d_in[1];
  const float* Wq    = (const float*)d_in[2];
  const float* Wqdw  = (const float*)d_in[3];
  const float* Wkv   = (const float*)d_in[4];
  const float* Wkvdw = (const float*)d_in[5];
  const float* Wproj = (const float*)d_in[6];
  const float* Wg1   = (const float*)d_in[7];
  const float* bg1   = (const float*)d_in[8];
  const float* Wg2   = (const float*)d_in[9];
  const float* bg2   = (const float*)d_in[10];
  const float* temp  = (const float*)d_in[11];
  float* out = (float*)d_out;

  char* base = (char*)d_ws;
  size_t off = 0;
  auto carve = [&](size_t bytes) -> char* {
    char* p = base + off;
    off += (bytes + 255) & ~(size_t)255;
    return p;
  };
  _Float16* q1   = (_Float16*)carve((size_t)BDIM * CDIM * SDIM * 2);      // pre-dw q
  _Float16* kv1  = (_Float16*)carve((size_t)BDIM * 2 * CDIM * SDIM * 2);  // pre-dw kv
  _Float16* qf   = (_Float16*)carve((size_t)BDIM * CDIM * SDIM * 2);      // q (post-dw)
  _Float16* kvf  = (_Float16*)carve((size_t)BDIM * 2 * CDIM * SDIM * 2);  // k|v
  _Float16* of   = (_Float16*)carve((size_t)BDIM * CDIM * SDIM * 2);      // attn out
  _Float16* g1   = (_Float16*)carve((size_t)BDIM * 192 * SDIM * 2);       // gate hidden
  float* invq    = (float*)carve((size_t)BDIM * CDIM * 4);
  float* invk    = (float*)carve((size_t)BDIM * CDIM * 4);
  float* attn    = (float*)carve((size_t)BDIM * HEADS * CPH * CPH * 4);
  _Float16* atb  = (_Float16*)carve((size_t)BDIM * HEADS * CPH * 64 * 2);
  float* partials= (float*)carve(128 * 4);
  int* dkp       = (int*)carve(4);

  dim3 blk(256);
  dim3 ggrid(SDIM / 64, BDIM);   // 64 N-strips x 8 batches
  // pointwise convs (GEMMs), activations read from HBM exactly once each
  scab_gemm1x1_fullk<float, _Float16, false><<<ggrid, blk, 0, stream>>>(
      Wq, x, nullptr, q1, CDIM);
  scab_gemm1x1_fullk<float, _Float16, false><<<ggrid, blk, 0, stream>>>(
      Wkv, y, nullptr, kv1, 2 * CDIM);
  scab_gemm1x1_fullk<float, _Float16, true><<<ggrid, blk, 0, stream>>>(
      Wg1, x, bg1, g1, 192);
  // depthwise 3x3
  scab_dwconv3x3<<<(BDIM * CDIM * SDIM) / 256, blk, 0, stream>>>(
      q1, Wqdw, qf, CDIM, BDIM * CDIM * SDIM);
  scab_dwconv3x3<<<(BDIM * 2 * CDIM * SDIM) / 256, blk, 0, stream>>>(
      kv1, Wkvdw, kvf, 2 * CDIM, BDIM * 2 * CDIM * SDIM);
  // L2 norms (q rows; k = first 384 channels of kv per batch)
  scab_rownorm<<<BDIM * CDIM, blk, 0, stream>>>(qf, CDIM, invq);
  scab_rownorm<<<BDIM * CDIM, blk, 0, stream>>>(kvf, 2 * CDIM, invk);
  // gate scalar -> dk
  scab_gate_partial<<<128, blk, 0, stream>>>(g1, Wg2, bg2, partials);
  scab_gate_finalize<<<1, 128, 0, stream>>>(partials, dkp);
  // attention
  scab_attn_qk<<<BDIM * HEADS, 288, 0, stream>>>(qf, kvf, invq, invk, temp, attn);
  scab_topk_softmax<<<BDIM * HEADS, 64, 0, stream>>>(attn, dkp, atb);
  scab_attn_v<<<dim3(32, BDIM * HEADS), blk, 0, stream>>>(atb, kvf, of);
  // output projection
  scab_gemm1x1_fullk<_Float16, float, false><<<ggrid, blk, 0, stream>>>(
      Wproj, of, nullptr, out, CDIM);
}
